// MultiHeadAttention_79998060855643
// MI455X (gfx1250) — compile-verified, matrix operand order checked by reading, not run on hardware
//
#include <hip/hip_runtime.h>
#include <hip/hip_bf16.h>

// ---------------------------------------------------------------------------
// MultiHeadAttention for MI455X (gfx1250, wave32, WMMA).
//   B=4, S=2048, D_MODEL=1024, H=16, HD=64.
// Strategy:
//   1. Convert inputs to f16; transpose weights to f16 (so WMMA B-fragments
//      are contiguous loads).
//   2. Q/K/V projections: WMMA f32_16x16x32_f16 GEMMs, 64x32 register-blocked
//      per wave (8 WMMAs per 6 fragment loads). Q pre-scaled by 1/8.
//   3. Flash-attention per (b, h, 32-query block): S^T = K*Q^T via WMMA,
//      online softmax with lane-local key reductions, O^T = V^T * P^T via
//      WMMA (P^T C-layout == B-operand layout, no shuffles). K/V fragments
//      reused across both 16-query tiles.
//   4. Output projection: WMMA GEMM, f32 accumulate + bias.
// ---------------------------------------------------------------------------

#define BATCH 4
#define SEQ 2048
#define DM 1024
#define NH 16
#define HD 64
#define ROWS (BATCH * SEQ)   // 8192

typedef __attribute__((ext_vector_type(16))) _Float16 v16h;
typedef __attribute__((ext_vector_type(8)))  _Float16 v8h;
typedef __attribute__((ext_vector_type(8)))  float    v8f;

__device__ __forceinline__ v8f wmma_f16(v16h a, v16h b, v8f c) {
  // D = A(16x32 f16) * B(32x16 f16) + C(16x16 f32)
  return __builtin_amdgcn_wmma_f32_16x16x32_f16(
      /*neg_a=*/false, a, /*neg_b=*/false, b,
      /*c_mod=*/(short)0, c, /*reuse_a=*/false, /*reuse_b=*/false);
}

// Per-lane A/B fragment loader for v_wmma_*_16x16x32_f16.
// Source is "row"-major with leading dim `ld` (halves). For the A operand,
// `row` is the M index; for the B operand, `row` is the N index (B columns
// are rows of a pre-transposed matrix). Lane L<16 holds k = [k0..k0+7,
// k0+16..k0+23]; lanes 16..31 hold k = [k0+8..k0+15, k0+24..k0+31].
__device__ __forceinline__ v16h load_frag(const _Float16* __restrict__ base,
                                          int row, int ld, int k0, int half) {
  const _Float16* p = base + (size_t)row * ld + k0 + 8 * half;
  v8h lo = *reinterpret_cast<const v8h*>(p);        // k0 + 8h      .. +7
  v8h hi = *reinterpret_cast<const v8h*>(p + 16);   // k0 + 16 + 8h .. +7
  v16h r;
#pragma unroll
  for (int i = 0; i < 8; ++i) { r[i] = lo[i]; r[i + 8] = hi[i]; }
  return r;
}

// --------------------------- data prep kernels -----------------------------

__global__ void cvt_f32_to_f16(const float* __restrict__ in,
                               _Float16* __restrict__ out, int n) {
  int i = blockIdx.x * blockDim.x + threadIdx.x;
  if (i < n) out[i] = (_Float16)in[i];
}

// [R][C] f32 -> [C][R] f16 (weight transpose; R=C=1024 here).
__global__ void transpose_w(const float* __restrict__ in,
                            _Float16* __restrict__ out, int R, int C) {
  int i = blockIdx.x * blockDim.x + threadIdx.x;
  if (i >= R * C) return;
  int r = i / C, c = i - r * C;
  out[(size_t)c * R + r] = (_Float16)in[i];
}

// V [B][S][H][HD] f16 -> Vt [B][H][HD][S] f16.
__global__ void transpose_v(const _Float16* __restrict__ v,
                            _Float16* __restrict__ vt) {
  int i = blockIdx.x * blockDim.x + threadIdx.x;  // B*S*H*HD = 2^23
  int d = i & (HD - 1);
  int h = (i >> 6) & (NH - 1);
  int s = (i >> 10) & (SEQ - 1);
  int b = i >> 21;
  vt[((((size_t)b * NH + h) * HD + d) * SEQ) + s] = v[i];
}

// ------------------------------ GEMM kernels -------------------------------
// One wave computes a 64(M) x 32(N) block: 4x2 accumulator tiles. Per k-step
// of 32: 4 A-fragments + 2 B-fragments feed 8 WMMAs (A reused across N tiles,
// B reused across M tiles).

// Y[m][n] = scale * sum_k X[m][k] * WT[n][k];  Y f16 [M][DM].
__global__ void gemm_proj(const _Float16* __restrict__ X,
                          const _Float16* __restrict__ WT,
                          _Float16* __restrict__ Y, int Kdim, float scale) {
  const int lane = threadIdx.x & 31;
  const int half = lane >> 4;
  const int lr = lane & 15;
  const int m0 = blockIdx.x * 64;
  const int n0 = blockIdx.y * 32;
  v8f acc[4][2];
#pragma unroll
  for (int i = 0; i < 4; ++i)
#pragma unroll
    for (int j = 0; j < 2; ++j) acc[i][j] = (v8f){};
  for (int k0 = 0; k0 < Kdim; k0 += 32) {
    v16h a[4], b[2];
#pragma unroll
    for (int i = 0; i < 4; ++i)
      a[i] = load_frag(X, m0 + 16 * i + lr, Kdim, k0, half);
#pragma unroll
    for (int j = 0; j < 2; ++j)
      b[j] = load_frag(WT, n0 + 16 * j + lr, Kdim, k0, half);
#pragma unroll
    for (int i = 0; i < 4; ++i)
#pragma unroll
      for (int j = 0; j < 2; ++j) acc[i][j] = wmma_f16(a[i], b[j], acc[i][j]);
  }
#pragma unroll
  for (int i = 0; i < 4; ++i)
#pragma unroll
    for (int j = 0; j < 2; ++j)
#pragma unroll
      for (int r = 0; r < 8; ++r) {
        int m = m0 + 16 * i + r + 8 * half;    // C-layout: lanes 16-31 -> M+8
        Y[(size_t)m * DM + n0 + 16 * j + lr] = (_Float16)(acc[i][j][r] * scale);
      }
}

// Out[m][n] = sum_k X[m][k] * WT[n][k] + bias[n];  f32 output.
__global__ void gemm_out(const _Float16* __restrict__ X,
                         const _Float16* __restrict__ WT,
                         const float* __restrict__ bias,
                         float* __restrict__ Y, int Kdim) {
  const int lane = threadIdx.x & 31;
  const int half = lane >> 4;
  const int lr = lane & 15;
  const int m0 = blockIdx.x * 64;
  const int n0 = blockIdx.y * 32;
  v8f acc[4][2];
#pragma unroll
  for (int i = 0; i < 4; ++i)
#pragma unroll
    for (int j = 0; j < 2; ++j) acc[i][j] = (v8f){};
  for (int k0 = 0; k0 < Kdim; k0 += 32) {
    v16h a[4], b[2];
#pragma unroll
    for (int i = 0; i < 4; ++i)
      a[i] = load_frag(X, m0 + 16 * i + lr, Kdim, k0, half);
#pragma unroll
    for (int j = 0; j < 2; ++j)
      b[j] = load_frag(WT, n0 + 16 * j + lr, Kdim, k0, half);
#pragma unroll
    for (int i = 0; i < 4; ++i)
#pragma unroll
      for (int j = 0; j < 2; ++j) acc[i][j] = wmma_f16(a[i], b[j], acc[i][j]);
  }
  float bn0 = bias[n0 + lr];
  float bn1 = bias[n0 + 16 + lr];
#pragma unroll
  for (int i = 0; i < 4; ++i)
#pragma unroll
    for (int r = 0; r < 8; ++r) {
      int m = m0 + 16 * i + r + 8 * half;
      Y[(size_t)m * DM + n0 + lr] = acc[i][0][r] + bn0;
      Y[(size_t)m * DM + n0 + 16 + lr] = acc[i][1][r] + bn1;
    }
}

// --------------------------- flash attention -------------------------------
// One wave handles a 32-query block (two 16-query tiles) for one (b, h).
// Computes S^T = K * Q^T so queries live in lanes and keys are lane-local
// (VGPR slot + half): softmax reductions are per-lane + one shfl_xor(16).
// P^T lands exactly in the B-operand layout for O^T = V^T * P^T. K and V^T
// fragments are loaded once and reused across both query tiles.
__global__ void attn_flash(const _Float16* __restrict__ Q,
                           const _Float16* __restrict__ K,
                           const _Float16* __restrict__ Vt,
                           _Float16* __restrict__ Ah) {
  const int lane = threadIdx.x & 31;
  const int half = lane >> 4;
  const int lr = lane & 15;
  const int qb0 = blockIdx.x * 32;    // query block base
  const int h  = blockIdx.y;
  const int b  = blockIdx.z;

  const size_t qkOff = (size_t)b * SEQ * DM + h * HD;
  const _Float16* qptr = Q + qkOff;
  const _Float16* kptr = K + qkOff;
  const _Float16* vt = Vt + ((size_t)b * NH + h) * HD * SEQ;

  // Q^T B-fragments (column = query = lr), held across the key loop.
  v16h qf[2][2];
#pragma unroll
  for (int q = 0; q < 2; ++q) {
    qf[q][0] = load_frag(qptr, qb0 + 16 * q + lr, DM, 0, half);
    qf[q][1] = load_frag(qptr, qb0 + 16 * q + lr, DM, 32, half);
  }

  v8f o[2][4];                        // O^T d-tiles 0..3 per query tile
#pragma unroll
  for (int q = 0; q < 2; ++q)
#pragma unroll
    for (int t = 0; t < 4; ++t) o[q][t] = (v8f){};
  float m_run[2] = {-1e30f, -1e30f};
  float l_run[2] = {0.0f, 0.0f};

  for (int kp = 0; kp < SEQ; kp += 32) {
    // K A-fragments, shared by both query tiles.
    v16h ak[4];
    ak[0] = load_frag(kptr, kp + lr, DM, 0, half);
    ak[1] = load_frag(kptr, kp + lr, DM, 32, half);
    ak[2] = load_frag(kptr, kp + 16 + lr, DM, 0, half);
    ak[3] = load_frag(kptr, kp + 16 + lr, DM, 32, half);

    v16h pf[2];
#pragma unroll
    for (int q = 0; q < 2; ++q) {
      // S^T tiles: s0 -> keys kp..kp+15, s1 -> keys kp+16..kp+31.
      v8f s0 = {}, s1 = {};
      s0 = wmma_f16(ak[0], qf[q][0], s0);
      s0 = wmma_f16(ak[1], qf[q][1], s0);
      s1 = wmma_f16(ak[2], qf[q][0], s1);
      s1 = wmma_f16(ak[3], qf[q][1], s1);
      // Online softmax over keys (lane-local slots + the opposite half).
      float mloc = -1e30f;
#pragma unroll
      for (int r = 0; r < 8; ++r) {
        mloc = fmaxf(mloc, s0[r]);
        mloc = fmaxf(mloc, s1[r]);
      }
      mloc = fmaxf(mloc, __shfl_xor(mloc, 16, 32));
      float m_new = fmaxf(m_run[q], mloc);
      float alpha = __expf(m_run[q] - m_new);
      float sl = 0.0f;
      v16h p;
#pragma unroll
      for (int r = 0; r < 8; ++r) {
        float e0 = __expf(s0[r] - m_new);
        float e1 = __expf(s1[r] - m_new);
        sl += e0 + e1;
        // P^T pack: lane<16 holds keys [0..7,16..23] == (tile0, tile1) slots;
        // lanes>=16 the complement. Exactly the B-operand layout.
        p[r] = (_Float16)e0;
        p[r + 8] = (_Float16)e1;
      }
      sl += __shfl_xor(sl, 16, 32);
      l_run[q] = l_run[q] * alpha + sl;
      m_run[q] = m_new;
#pragma unroll
      for (int t = 0; t < 4; ++t)
#pragma unroll
        for (int r = 0; r < 8; ++r) o[q][t][r] *= alpha;
      pf[q] = p;
    }

    // O^T += V^T * P^T; V^T fragments shared by both query tiles.
#pragma unroll
    for (int t = 0; t < 4; ++t) {
      v16h av = load_frag(vt, 16 * t + lr, SEQ, kp, half);
      o[0][t] = wmma_f16(av, pf[0], o[0][t]);
      o[1][t] = wmma_f16(av, pf[1], o[1][t]);
    }
  }

  // O^T tile t: lane holds query q=lr, d = 16*t + 8*half + r (r contiguous).
#pragma unroll
  for (int q = 0; q < 2; ++q) {
    const float inv = 1.0f / l_run[q];
    _Float16* orow =
        Ah + ((size_t)b * SEQ + qb0 + 16 * q + lr) * DM + h * HD + 8 * half;
#pragma unroll
    for (int t = 0; t < 4; ++t) {
      v8h w;
#pragma unroll
      for (int r = 0; r < 8; ++r) w[r] = (_Float16)(o[q][t][r] * inv);
      *reinterpret_cast<v8h*>(orow + 16 * t) = w;
    }
  }
}

// ------------------------------- launcher ----------------------------------

extern "C" void kernel_launch(void* const* d_in, const int* in_sizes, int n_in,
                              void* d_out, int out_size, void* d_ws, size_t ws_size,
                              hipStream_t stream) {
  const float* x  = (const float*)d_in[0];  // [B,S,DM]
  const float* Wq = (const float*)d_in[1];  // [DM, H, HD] == [1024][1024]
  const float* Wk = (const float*)d_in[3];
  const float* Wv = (const float*)d_in[5];
  const float* Wo = (const float*)d_in[7];  // [H, HD, DM] == [1024][1024]
  const float* bo = (const float*)d_in[8];  // [DM]
  float* out = (float*)d_out;               // [B,S,DM] f32

  // Workspace carve-up (halves).
  const size_t NMAT = (size_t)ROWS * DM;    // 8192*1024
  _Float16* Xh  = (_Float16*)d_ws;
  _Float16* Qh  = Xh + NMAT;
  _Float16* Kh  = Qh + NMAT;
  _Float16* Vh  = Kh + NMAT;
  _Float16* Vt  = Vh + NMAT;
  _Float16* Ah  = Vt + NMAT;
  _Float16* WqT = Ah + NMAT;
  _Float16* WkT = WqT + (size_t)DM * DM;
  _Float16* WvT = WkT + (size_t)DM * DM;
  _Float16* WoT = WvT + (size_t)DM * DM;

  // 1) f16 conversions / weight transposes.
  {
    int n = ROWS * DM;
    cvt_f32_to_f16<<<(n + 255) / 256, 256, 0, stream>>>(x, Xh, n);
    int nw = DM * DM;
    transpose_w<<<(nw + 255) / 256, 256, 0, stream>>>(Wq, WqT, DM, DM);
    transpose_w<<<(nw + 255) / 256, 256, 0, stream>>>(Wk, WkT, DM, DM);
    transpose_w<<<(nw + 255) / 256, 256, 0, stream>>>(Wv, WvT, DM, DM);
    transpose_w<<<(nw + 255) / 256, 256, 0, stream>>>(Wo, WoT, DM, DM);
  }

  // 2) Q/K/V projections (1/sqrt(HD)=0.125 folded into Q).
  {
    dim3 grid(ROWS / 64, DM / 32);
    gemm_proj<<<grid, 32, 0, stream>>>(Xh, WqT, Qh, DM, 0.125f);
    gemm_proj<<<grid, 32, 0, stream>>>(Xh, WkT, Kh, DM, 1.0f);
    gemm_proj<<<grid, 32, 0, stream>>>(Xh, WvT, Vh, DM, 1.0f);
  }

  // 3) V transpose for contiguous V^T A-fragments.
  {
    int n = ROWS * DM;
    transpose_v<<<(n + 255) / 256, 256, 0, stream>>>(Vh, Vt);
  }

  // 4) Flash attention: one wave per (b, h, 32-query block).
  {
    dim3 grid(SEQ / 32, NH, BATCH);
    attn_flash<<<grid, 32, 0, stream>>>(Qh, Kh, Vt, Ah);
  }

  // 5) Output projection + bias.
  {
    dim3 grid(ROWS / 64, DM / 32);
    gemm_out<<<grid, 32, 0, stream>>>(Ah, WoT, bo, out, DM);
  }
}
